// SegSmall_3642132267194
// MI455X (gfx1250) — matrix operand order repaired and checked by reading, not compiled, not run on hardware
//
#include <hip/hip_runtime.h>
#include <hip/hip_bf16.h>

typedef __attribute__((ext_vector_type(16))) _Float16 v16h;
typedef __attribute__((ext_vector_type(8)))  _Float16 v8h;
typedef __attribute__((ext_vector_type(8)))  float    v8f;

static constexpr int BATCH = 8;

static inline int ceildiv_h(int a, int b) { return (a + b - 1) / b; }

// ---------------------------------------------------------------------------
// KNN: blocks assigned per (batch, 256-query chunk); source points stream
// through LDS tiles so the O(Mq*Ns) scan reads each source point once per
// 256 queries instead of once per query. Top-K kept via register insertion.
// ---------------------------------------------------------------------------
template<int K>
__global__ void knn_kernel(const float* __restrict__ qpts,
                           const float* __restrict__ spts,
                           int* __restrict__ out_idx,
                           int Mq, int Ns, int qblocks)
{
    __shared__ float sx[256], sy[256], sz[256];
    int b  = blockIdx.x / qblocks;
    int m  = (blockIdx.x % qblocks) * 256 + threadIdx.x;
    bool valid = m < Mq;
    int t = b * Mq + (valid ? m : 0);

    const float* q = qpts + (size_t)t * 3;
    float qx = q[0], qy = q[1], qz = q[2];
    float bd[K]; int bi[K];
#pragma unroll
    for (int i = 0; i < K; ++i) { bd[i] = 3.0e38f; bi[i] = 0; }

    const float* sp = spts + (size_t)b * Ns * 3;
    for (int base = 0; base < Ns; base += 256) {
        int chunk = min(256, Ns - base);
        if (threadIdx.x < (unsigned)chunk) {
            const float* p = sp + (size_t)(base + threadIdx.x) * 3;
            sx[threadIdx.x] = p[0]; sy[threadIdx.x] = p[1]; sz[threadIdx.x] = p[2];
        }
        __syncthreads();
        for (int j = 0; j < chunk; ++j) {
            float dx = sx[j] - qx, dy = sy[j] - qy, dz = sz[j] - qz;
            float d  = dx*dx + dy*dy + dz*dz;
            if (d < bd[K-1]) {
                bd[K-1] = d; bi[K-1] = base + j;
#pragma unroll
                for (int i = K-1; i > 0; --i) {
                    if (bd[i] < bd[i-1]) {
                        float td = bd[i]; bd[i] = bd[i-1]; bd[i-1] = td;
                        int ti = bi[i];  bi[i] = bi[i-1];  bi[i-1] = ti;
                    }
                }
            }
        }
        __syncthreads();
    }
    if (valid) {
#pragma unroll
        for (int i = 0; i < K; ++i) out_idx[(size_t)t * K + i] = bi[i];
    }
}

// ---------------------------------------------------------------------------
// rel features -> f16 MLP input, row stride 64 (cols 48..63 zeroed pad)
// ---------------------------------------------------------------------------
template<int K>
__global__ void relfeat_kernel(const float* __restrict__ spts,
                               const float* __restrict__ qpts,
                               const int* __restrict__ idx,
                               const float* __restrict__ centers, // 3 x 16
                               _Float16* __restrict__ dfeat,      // (B*Mq*K) x 64
                               int Mq, int Ns)
{
    int t = blockIdx.x * blockDim.x + threadIdx.x;
    if (t >= BATCH * Mq) return;
    int b = t / Mq;
    const float* q = qpts + (size_t)t * 3;
    float qx = q[0], qy = q[1], qz = q[2];
    float rx[K], ry[K], rz[K];
    float m2 = 0.0f;
#pragma unroll
    for (int k = 0; k < K; ++k) {
        int j = idx[(size_t)t * K + k];
        const float* sp = spts + ((size_t)b * Ns + j) * 3;
        rx[k] = sp[0] - qx; ry[k] = sp[1] - qy; rz[k] = sp[2] - qz;
        m2 = fmaxf(m2, rx[k]*rx[k] + ry[k]*ry[k] + rz[k]*rz[k]);
    }
    float maxi = sqrtf(m2);
    float inv  = (maxi == 0.0f) ? 1.0f : 1.0f / maxi;
#pragma unroll
    for (int k = 0; k < K; ++k) {
        _Float16* d = dfeat + ((size_t)t * K + k) * 64;
        float r0 = rx[k]*inv, r1 = ry[k]*inv, r2 = rz[k]*inv;
        for (int c = 0; c < 16; ++c) {
            d[c]      = (_Float16)(r0 - centers[c]);
            d[16 + c] = (_Float16)(r1 - centers[16 + c]);
            d[32 + c] = (_Float16)(r2 - centers[32 + c]);
            d[48 + c] = (_Float16)0.0f;   // K padding
        }
    }
}

// ---------------------------------------------------------------------------
// Weight prep: Bt[n][k] = f16(W[k][n]), zero-padded to Npad x Kpad.
// Run once per GEMM; lets the GEMM read B fragments straight from global.
// ---------------------------------------------------------------------------
__global__ void wprep_kernel(const float* __restrict__ W, // Kd x N row-major
                             _Float16* __restrict__ Bt,   // Npad x Kpad
                             int Kd, int N, int Kpad, int Npad)
{
    int t = blockIdx.x * blockDim.x + threadIdx.x;
    if (t >= Npad * Kpad) return;
    int n = t / Kpad, k = t % Kpad;
    float v = (n < N && k < Kd) ? W[(size_t)k * N + n] : 0.0f;
    Bt[t] = (_Float16)v;
}

// ---------------------------------------------------------------------------
// LDS-free WMMA GEMM: Out[M,Npad] = act(A[M,Kpad] @ Bt^T * scale + bias)
// 8 waves per block, one 16x16 C tile per wave. A (lda halves, 16B aligned)
// and pre-transposed Bt are loaded directly as global_load_b128 fragments in
// the exact gfx1250 per-lane WMMA layouts. No LDS, no barriers.
// ---------------------------------------------------------------------------
template<bool RELU, bool OUT16>
__global__ void gemm_wmma_kernel(const _Float16* __restrict__ A, int lda,
                                 const _Float16* __restrict__ Bt, // Npad x Kpad
                                 const float* __restrict__ bias, int Nreal,
                                 float scale,
                                 void* __restrict__ Out, int ldo,
                                 int M, int Kpad)
{
    int wave = threadIdx.x >> 5;
    int lane = threadIdx.x & 31;
    int wm = (blockIdx.x * 8 + wave) * 16;
    if (wm >= M) return;                       // wave-uniform
    int n0 = blockIdx.y * 16;
    int hl = lane & 15, hi = lane >> 4;

    const _Float16* arow = A  + (size_t)(wm + hl) * lda;
    const _Float16* brow = Bt + (size_t)(n0 + hl) * Kpad;

    v8f acc = {};
    int nK = Kpad >> 5;
    for (int kc = 0; kc < nK; ++kc) {
        int k0 = kc << 5;
        // A 16-bit 16x32 layout: lane group hi holds K = hi*8..+7, 16+hi*8..+7
        v8h a_lo = *(const v8h*)(arow + k0 + hi * 8);
        v8h a_hi = *(const v8h*)(arow + k0 + 16 + hi * 8);
        // B 16-bit 32x16 layout: lane group hi holds K = hi*8..+7, 16+hi*8..+7
        v8h b_lo = *(const v8h*)(brow + k0 + hi * 8);
        v8h b_hi = *(const v8h*)(brow + k0 + 16 + hi * 8);
        if (kc + 1 < nK) __builtin_prefetch(arow + k0 + 32, 0, 3);
        v16h av = __builtin_shufflevector(a_lo, a_hi,
                    0,1,2,3,4,5,6,7,8,9,10,11,12,13,14,15);
        v16h bv = __builtin_shufflevector(b_lo, b_hi,
                    0,1,2,3,4,5,6,7,8,9,10,11,12,13,14,15);
        acc = __builtin_amdgcn_wmma_f32_16x16x32_f16(false, av, false, bv,
                                                     (short)0, acc, false, false);
    }

    // C/D layout: VGPR r -> row = r + hi*8, col = lane&15
    int col = n0 + hl;
    float badd = (bias != nullptr && col < Nreal) ? bias[col] : 0.0f;
#pragma unroll
    for (int r = 0; r < 8; ++r) {
        int row = wm + r + hi * 8;
        float v = acc[r] * scale + badd;
        if (RELU) v = fmaxf(v, 0.0f);
        if (OUT16) ((_Float16*)Out)[(size_t)row * ldo + col] = (_Float16)v;
        else       ((float*)Out)[(size_t)row * ldo + col]    = v;
    }
}

// ---------------------------------------------------------------------------
// Aggregation: agg[bm, c*16+n] = sum_k feats[b, idx[bm,k], c] * h[bm*K+k, n]
// Writes the full padded row (Epad columns): zeros for c >= cin.
// ---------------------------------------------------------------------------
__global__ void agg_kernel(const float* __restrict__ feats,
                           const int* __restrict__ idx,
                           const _Float16* __restrict__ h, int ldh,
                           _Float16* __restrict__ agg,
                           int Md, int Ns, int cin, int K, int Epad)
{
    int bm = blockIdx.x;            // 0 .. B*Md-1
    int b  = bm / Md;
    __shared__ float sh[16 * 16];
    __shared__ int   sidx[16];
    for (int e = threadIdx.x; e < K * 16; e += blockDim.x)
        sh[e] = (float)h[(size_t)(bm * K + (e >> 4)) * ldh + (e & 15)];
    for (int e = threadIdx.x; e < K; e += blockDim.x)
        sidx[e] = idx[(size_t)bm * K + e];
    __syncthreads();
    int E = cin * 16;
    for (int e = threadIdx.x; e < Epad; e += blockDim.x) {
        float sum = 0.0f;
        if (e < E) {
            int c = e >> 4, n = e & 15;
            for (int k = 0; k < K; ++k)
                sum += feats[((size_t)b * Ns + sidx[k]) * cin + c] * sh[k * 16 + n];
        }
        agg[(size_t)bm * Epad + e] = (_Float16)sum;
    }
}

// ---------------------------------------------------------------------------
// BatchNorm over rows (one block per channel) + fused ReLU apply into a
// possibly concatenated destination slice.
// ---------------------------------------------------------------------------
__global__ void bn_stats_kernel(const float* __restrict__ x,
                                float* __restrict__ stats, int rows, int C)
{
    int c = blockIdx.x;
    float s = 0.0f, s2 = 0.0f;
    for (int r = threadIdx.x; r < rows; r += blockDim.x) {
        float v = x[(size_t)r * C + c];
        s += v; s2 += v * v;
    }
    __shared__ float red[256], red2[256];
    red[threadIdx.x] = s; red2[threadIdx.x] = s2;
    __syncthreads();
    for (int off = 128; off > 0; off >>= 1) {
        if (threadIdx.x < (unsigned)off) {
            red[threadIdx.x]  += red[threadIdx.x + off];
            red2[threadIdx.x] += red2[threadIdx.x + off];
        }
        __syncthreads();
    }
    if (threadIdx.x == 0) {
        float mean = red[0] / (float)rows;
        stats[c]     = mean;
        stats[C + c] = red2[0] / (float)rows - mean * mean;
    }
}

__global__ void bn_apply_kernel(const float* __restrict__ x,
                                const float* __restrict__ stats,
                                const float* __restrict__ gamma,
                                const float* __restrict__ beta,
                                float* __restrict__ dst,
                                int rows, int C, int dstStride, int chOff)
{
    int t = blockIdx.x * blockDim.x + threadIdx.x;
    if (t >= rows * C) return;
    int r = t / C, c = t % C;
    float v = (x[t] - stats[c]) * rsqrtf(stats[C + c] + 1e-5f) * gamma[c] + beta[c];
    dst[(size_t)r * dstStride + chOff + c] = fmaxf(v, 0.0f);
}

__global__ void copy_ch_kernel(const float* __restrict__ src,
                               float* __restrict__ dst,
                               int rows, int C, int dstStride, int chOff)
{
    int t = blockIdx.x * blockDim.x + threadIdx.x;
    if (t >= rows * C) return;
    int r = t / C, c = t % C;
    dst[(size_t)r * dstStride + chOff + c] = src[(size_t)r * C + c];
}

__global__ void subset_pts_kernel(const float* __restrict__ src,
                                  float* __restrict__ dst, int Nsrc, int Ndst)
{
    int t = blockIdx.x * blockDim.x + threadIdx.x;
    if (t >= BATCH * Ndst * 3) return;
    int b = t / (Ndst * 3), rem = t % (Ndst * 3);
    dst[t] = src[(size_t)b * Nsrc * 3 + rem];
}

__global__ void fc_kernel(const float* __restrict__ x,
                          const float* __restrict__ w,   // 48 x 13
                          const float* __restrict__ b,
                          float* __restrict__ out, int rows)
{
    int t = blockIdx.x * blockDim.x + threadIdx.x;
    if (t >= rows * 13) return;
    int r = t / 13, o = t % 13;
    float s = b[o];
#pragma unroll
    for (int c = 0; c < 48; ++c) s += x[(size_t)r * 48 + c] * w[c * 13 + o];
    out[t] = s;
}

// ---------------------------------------------------------------------------
// Host-side orchestration
// ---------------------------------------------------------------------------
struct ConvP {
    const float *weight, *centers, *l1w, *l1b, *l2w, *l2b, *l3w, *l3b;
};

static void launch_gemm(hipStream_t st, const _Float16* A, int lda,
                        const _Float16* Bt, const float* bias, int Nreal,
                        float scale, void* out, int ldo,
                        int M, int Kpad, int Npad, bool relu_out16)
{
    dim3 grid(ceildiv_h(M, 128), Npad / 16), blk(256);
    if (relu_out16)
        gemm_wmma_kernel<true, true><<<grid, blk, 0, st>>>(A, lda, Bt, bias, Nreal,
                                                           scale, out, ldo, M, Kpad);
    else
        gemm_wmma_kernel<false, false><<<grid, blk, 0, st>>>(A, lda, Bt, bias, Nreal,
                                                             scale, out, ldo, M, Kpad);
}

static void run_layer(hipStream_t st,
                      const float* feats, int cin, int Ns,
                      const float* ptsS, const float* ptsD, int Md, int K, int co,
                      const ConvP& cp, const float* gamma, const float* beta,
                      float* dst, int dstStride, int chOff,
                      char* scratch, float* stats)
{
    int    BMd = BATCH * Md;
    size_t BMK = (size_t)BMd * K;
    int    Kp  = cin * 16;        // projection K
    int    Kpp = (Kp + 31) & ~31; // padded to multiple of 32
    char* cur = scratch;
    auto carve = [&](size_t bytes) -> char* {
        char* p = cur; cur += (bytes + 255) & ~(size_t)255; return p;
    };
    int*      idx = (int*)      carve(BMK * 4);
    _Float16* dft = (_Float16*) carve(BMK * 64 * 2);
    _Float16* h1  = (_Float16*) carve(BMK * 32 * 2);
    _Float16* h2  = (_Float16*) carve(BMK * 32 * 2);
    _Float16* h3  = (_Float16*) carve(BMK * 32 * 2);
    _Float16* agg = (_Float16*) carve((size_t)BMd * Kpp * 2);
    float*    pre = (float*)    carve((size_t)BMd * co * 4);
    _Float16* w1t = (_Float16*) carve((size_t)32 * 64 * 2);
    _Float16* w2t = (_Float16*) carve((size_t)32 * 32 * 2);
    _Float16* w3t = (_Float16*) carve((size_t)32 * 32 * 2);
    _Float16* wot = (_Float16*) carve((size_t)co * Kpp * 2);

    // weight prep (transpose + f16 + pad)
    wprep_kernel<<<ceildiv_h(32 * 64, 256), 256, 0, st>>>(cp.l1w, w1t, 48, 32, 64, 32);
    wprep_kernel<<<ceildiv_h(32 * 32, 256), 256, 0, st>>>(cp.l2w, w2t, 32, 16, 32, 32);
    wprep_kernel<<<ceildiv_h(32 * 32, 256), 256, 0, st>>>(cp.l3w, w3t, 16, 16, 32, 32);
    wprep_kernel<<<ceildiv_h(co * Kpp, 256), 256, 0, st>>>(cp.weight, wot, Kp, co, Kpp, co);

    int qblocks = ceildiv_h(Md, 256);
    dim3 kgrid(BATCH * qblocks), kblk(256);
    int nq = ceildiv_h(BMd, 256);
    switch (K) {
    case 4:
        knn_kernel<4><<<kgrid, kblk, 0, st>>>(ptsD, ptsS, idx, Md, Ns, qblocks);
        relfeat_kernel<4><<<nq, 256, 0, st>>>(ptsS, ptsD, idx, cp.centers, dft, Md, Ns);
        break;
    case 8:
        knn_kernel<8><<<kgrid, kblk, 0, st>>>(ptsD, ptsS, idx, Md, Ns, qblocks);
        relfeat_kernel<8><<<nq, 256, 0, st>>>(ptsS, ptsD, idx, cp.centers, dft, Md, Ns);
        break;
    default:
        knn_kernel<16><<<kgrid, kblk, 0, st>>>(ptsD, ptsS, idx, Md, Ns, qblocks);
        relfeat_kernel<16><<<nq, 256, 0, st>>>(ptsS, ptsD, idx, cp.centers, dft, Md, Ns);
        break;
    }
    // per-neighbor MLP 48->32->16->16, all LDS-free WMMA, f16 activations
    launch_gemm(st, dft, 64, w1t, cp.l1b, 32, 1.0f, h1, 32, (int)BMK, 64, 32, true);
    launch_gemm(st, h1,  32, w2t, cp.l2b, 16, 1.0f, h2, 32, (int)BMK, 32, 32, true);
    launch_gemm(st, h2,  32, w3t, cp.l3b, 16, 1.0f, h3, 32, (int)BMK, 32, 32, true);
    // neighbor aggregation -> (BMd, Kpp) with zeroed K padding
    agg_kernel<<<BMd, 256, 0, st>>>(feats, idx, h3, 32, agg, Md, Ns, cin, K, Kpp);
    // projection (cin*16 -> co), /K ; then BN + ReLU into destination slice
    launch_gemm(st, agg, Kpp, wot, nullptr, co, 1.0f / (float)K, pre, co, BMd, Kpp, co, false);
    bn_stats_kernel<<<co, 256, 0, st>>>(pre, stats, BMd, co);
    bn_apply_kernel<<<ceildiv_h(BMd * co, 256), 256, 0, st>>>(pre, stats, gamma, beta,
                                                              dst, BMd, co, dstStride, chOff);
}

extern "C" void kernel_launch(void* const* d_in, const int* in_sizes, int n_in,
                              void* d_out, int out_size, void* d_ws, size_t ws_size,
                              hipStream_t stream)
{
    (void)in_sizes; (void)n_in; (void)out_size; (void)ws_size;
    auto in = [&](int i) { return (const float*)d_in[i]; };

    const float* x    = in(0);   // (8, 8192, 3)
    const float* pts1 = in(1);   // (8, 8192, 3)

    auto conv = [&](int i) {
        ConvP c; int b = 2 + i * 8;
        c.weight = in(b); c.centers = in(b + 1);
        c.l1w = in(b + 2); c.l1b = in(b + 3);
        c.l2w = in(b + 4); c.l2b = in(b + 5);
        c.l3w = in(b + 6); c.l3b = in(b + 7);
        return c;
    };
    auto bng = [&](int i) { return in(82 + 2 * i); };
    auto bnb = [&](int i) { return in(83 + 2 * i); };
    const float* fcw = in(102);
    const float* fcb = in(103);

    char* w = (char*)d_ws;
    auto alloc = [&](size_t bytes) -> char* {
        char* p = w; w += (bytes + 255) & ~(size_t)255; return p;
    };
    float* pts2 = (float*)alloc((size_t)BATCH * 1024 * 3 * 4);
    float* pts3 = (float*)alloc((size_t)BATCH * 256 * 3 * 4);
    float* pts4 = (float*)alloc((size_t)BATCH * 64 * 3 * 4);
    float* pts5 = (float*)alloc((size_t)BATCH * 16 * 3 * 4);
    float* pts6 = (float*)alloc((size_t)BATCH * 8 * 3 * 4);
    float* x2   = (float*)alloc((size_t)BATCH * 1024 * 48 * 4);
    float* x3   = (float*)alloc((size_t)BATCH * 256 * 48 * 4);
    float* x4   = (float*)alloc((size_t)BATCH * 64 * 96 * 4);
    float* x5   = (float*)alloc((size_t)BATCH * 16 * 96 * 4);
    float* x6   = (float*)alloc((size_t)BATCH * 8 * 96 * 4);
    float* x5dc = (float*)alloc((size_t)BATCH * 16 * 192 * 4);
    float* x4dc = (float*)alloc((size_t)BATCH * 64 * 192 * 4);
    float* x3dc = (float*)alloc((size_t)BATCH * 256 * 96 * 4);
    float* x2dc = (float*)alloc((size_t)BATCH * 1024 * 96 * 4);
    float* x1d  = (float*)alloc((size_t)BATCH * 8192 * 48 * 4);
    float* stats = (float*)alloc(2 * 192 * 4);
    char*  scratch = w;

    subset_pts_kernel<<<ceildiv_h(BATCH * 1024 * 3, 256), 256, 0, stream>>>(pts1, pts2, 8192, 1024);
    subset_pts_kernel<<<ceildiv_h(BATCH * 256 * 3, 256), 256, 0, stream>>>(pts1, pts3, 8192, 256);
    subset_pts_kernel<<<ceildiv_h(BATCH * 64 * 3, 256), 256, 0, stream>>>(pts1, pts4, 8192, 64);
    subset_pts_kernel<<<ceildiv_h(BATCH * 16 * 3, 256), 256, 0, stream>>>(pts1, pts5, 8192, 16);
    subset_pts_kernel<<<ceildiv_h(BATCH * 8 * 3, 256), 256, 0, stream>>>(pts1, pts6, 8192, 8);

    run_layer(stream, x,    3,  8192, pts1, pts2, 1024, 16, 48, conv(0), bng(0), bnb(0), x2,  48, 0, scratch, stats);
    run_layer(stream, x2,  48,  1024, pts2, pts3,  256, 16, 48, conv(1), bng(1), bnb(1), x3,  48, 0, scratch, stats);
    run_layer(stream, x3,  48,   256, pts3, pts4,   64,  8, 96, conv(2), bng(2), bnb(2), x4,  96, 0, scratch, stats);
    run_layer(stream, x4,  96,    64, pts4, pts5,   16,  8, 96, conv(3), bng(3), bnb(3), x5,  96, 0, scratch, stats);
    run_layer(stream, x5,  96,    16, pts5, pts6,    8,  4, 96, conv(4), bng(4), bnb(4), x6,  96, 0, scratch, stats);
    run_layer(stream, x6,  96,     8, pts6, pts5,   16,  4, 96, conv(5), bng(5), bnb(5), x5dc, 192, 0, scratch, stats);
    copy_ch_kernel<<<ceildiv_h(BATCH * 16 * 96, 256), 256, 0, stream>>>(x5, x5dc, BATCH * 16, 96, 192, 96);
    run_layer(stream, x5dc, 192,  16, pts5, pts4,   64,  4, 96, conv(6), bng(6), bnb(6), x4dc, 192, 0, scratch, stats);
    copy_ch_kernel<<<ceildiv_h(BATCH * 64 * 96, 256), 256, 0, stream>>>(x4, x4dc, BATCH * 64, 96, 192, 96);
    run_layer(stream, x4dc, 192,  64, pts4, pts3,  256,  4, 48, conv(7), bng(7), bnb(7), x3dc, 96, 0, scratch, stats);
    copy_ch_kernel<<<ceildiv_h(BATCH * 256 * 48, 256), 256, 0, stream>>>(x3, x3dc, BATCH * 256, 48, 96, 48);
    run_layer(stream, x3dc, 96,  256, pts3, pts2, 1024,  8, 48, conv(8), bng(8), bnb(8), x2dc, 96, 0, scratch, stats);
    copy_ch_kernel<<<ceildiv_h(BATCH * 1024 * 48, 256), 256, 0, stream>>>(x2, x2dc, BATCH * 1024, 48, 96, 48);
    run_layer(stream, x2dc, 96, 1024, pts2, pts1, 8192,  8, 48, conv(9), bng(9), bnb(9), x1d, 48, 0, scratch, stats);

    fc_kernel<<<ceildiv_h(BATCH * 8192 * 13, 256), 256, 0, stream>>>(x1d, fcw, fcb,
                                                                     (float*)d_out, BATCH * 8192);
}
